// SelfAttention_xyz_2216203124948
// MI455X (gfx1250) — compile-verified, matrix operand order checked by reading, not run on hardware
//
#include <hip/hip_runtime.h>

typedef __attribute__((ext_vector_type(16))) __bf16 v16bf;
typedef __attribute__((ext_vector_type(8)))  float  v8f;

union FragAB {
    v16bf v;
    uint4 q[2];
};

// ---------------------------------------------------------------------------
// fp32 -> bf16 conversion (grid-stride)
// ---------------------------------------------------------------------------
__global__ void cvt_f32_to_bf16(const float* __restrict__ in,
                                __bf16* __restrict__ out, int n) {
    int i = blockIdx.x * blockDim.x + threadIdx.x;
    int stride = gridDim.x * blockDim.x;
    for (; i < n; i += stride) out[i] = (__bf16)in[i];
}

// ---------------------------------------------------------------------------
// Fused QKV projection: out = x @ W^T + b   (M=8192, N=D=1024, K=D=1024)
// grid = (N/128, M/256, 3)   z: 0=Q (scaled 1/32), 1=K, 2=V (written transposed)
// 8 waves/block, block tile 256x128, wave tile 64x64 = 4x4 WMMA accumulators:
// 16 wmma per k-step vs 16 ds_load_b128 (1:1 compute:LDS-fragment ratio).
// ---------------------------------------------------------------------------
__global__ __launch_bounds__(256) void qkv_gemm_wmma(
    const __bf16* __restrict__ Xb,
    const __bf16* __restrict__ Wqb, const __bf16* __restrict__ Wkb,
    const __bf16* __restrict__ Wvb,
    const float* __restrict__ bq, const float* __restrict__ bk,
    const float* __restrict__ bv,
    __bf16* __restrict__ Q, __bf16* __restrict__ Kc, __bf16* __restrict__ Vt) {
    const int D = 1024, S = 2048;
    __shared__ __bf16 As[256][40];  // 80B pitch: 16B aligned, bank-stagger
    __shared__ __bf16 Bs[128][40];

    const int tid  = threadIdx.x;
    const int z    = blockIdx.z;
    const __bf16* W    = (z == 0) ? Wqb : (z == 1) ? Wkb : Wvb;
    const float*  bias = (z == 0) ? bq  : (z == 1) ? bk  : bv;

    const int n0 = blockIdx.x * 128;
    const int m0 = blockIdx.y * 256;
    const int lane = tid & 31;
    const int wid  = tid >> 5;
    const int wm   = (wid & 3) * 64;   // 4 row-groups of 64
    const int wn   = (wid >> 2) * 64;  // 2 col-groups of 64
    const int lrow = lane & 15;
    const int off  = (lane >> 4) * 8;  // K-half select per ISA A-layout

    v8f acc[4][4] = {};

    for (int kk = 0; kk < D; kk += 32) {
#pragma unroll
        for (int i = 0; i < 4; i++) {   // A tile: 256x32 -> 1024 16B chunks
            int c  = tid + (i << 8);
            int r  = c >> 2;
            int kc = (c & 3) << 3;
            *reinterpret_cast<uint4*>(&As[r][kc]) =
                *reinterpret_cast<const uint4*>(Xb + (size_t)(m0 + r) * D + kk + kc);
        }
#pragma unroll
        for (int i = 0; i < 2; i++) {   // B tile: 128x32 -> 512 16B chunks
            int c  = tid + (i << 8);
            int r  = c >> 2;
            int kc = (c & 3) << 3;
            *reinterpret_cast<uint4*>(&Bs[r][kc]) =
                *reinterpret_cast<const uint4*>(W + (size_t)(n0 + r) * D + kk + kc);
        }
        if (kk + 32 < D)  // speculative prefetch of next A tile slice
            __builtin_prefetch(Xb + (size_t)(m0 + tid) * D + kk + 32, 0, 1);
        __syncthreads();

        FragAB a[4], b[4];
#pragma unroll
        for (int i = 0; i < 4; i++) {
            int r = wm + i * 16 + lrow;
            a[i].q[0] = *reinterpret_cast<const uint4*>(&As[r][off]);
            a[i].q[1] = *reinterpret_cast<const uint4*>(&As[r][off + 16]);
        }
#pragma unroll
        for (int j = 0; j < 4; j++) {
            int r = wn + j * 16 + lrow;
            b[j].q[0] = *reinterpret_cast<const uint4*>(&Bs[r][off]);
            b[j].q[1] = *reinterpret_cast<const uint4*>(&Bs[r][off + 16]);
        }
#pragma unroll
        for (int i = 0; i < 4; i++)
#pragma unroll
            for (int j = 0; j < 4; j++)
                acc[i][j] = __builtin_amdgcn_wmma_f32_16x16x32_bf16(
                    false, a[i].v, false, b[j].v, (short)0, acc[i][j], false, false);
        __syncthreads();
    }

    const float scale = (z == 0) ? 0.03125f : 1.0f;  // 1/sqrt(1024) folded into Q
#pragma unroll
    for (int i = 0; i < 4; i++) {
#pragma unroll
        for (int j = 0; j < 4; j++) {
#pragma unroll
            for (int v = 0; v < 8; v++) {
                int gm = m0 + wm + i * 16 + v + ((lane >> 4) << 3);
                int gn = n0 + wn + j * 16 + (lane & 15);
                float cv = (acc[i][j][v] + bias[gn]) * scale;
                if (z == 2) {
                    int bb = gm >> 11, ss = gm & 2047;
                    Vt[((size_t)bb * D + gn) * S + ss] = (__bf16)cv;
                } else {
                    __bf16* O = (z == 0) ? Q : Kc;
                    O[(size_t)gm * D + gn] = (__bf16)cv;
                }
            }
        }
    }
}

// ---------------------------------------------------------------------------
// Generic batched GEMM: C = A (MxK) * Bt^T, Bt stored row-major NxK (ld = K).
// grid = (N/128, M/256, batches). out_bf16 selects bf16 vs f32 output.
// Same 256x128 block tile / 64x64 wave tile as above.
// ---------------------------------------------------------------------------
__global__ __launch_bounds__(256) void gemm_bt_wmma(
    const __bf16* __restrict__ A, const __bf16* __restrict__ Bt,
    void* __restrict__ Out, int M, int N, int Kd,
    long long sA, long long sB, long long sO, int out_bf16) {
    __shared__ __bf16 As[256][40];
    __shared__ __bf16 Bs[128][40];

    A  += (size_t)blockIdx.z * sA;
    Bt += (size_t)blockIdx.z * sB;
    const size_t ob = (size_t)blockIdx.z * sO;

    const int tid  = threadIdx.x;
    const int n0   = blockIdx.x * 128;
    const int m0   = blockIdx.y * 256;
    const int lane = tid & 31;
    const int wid  = tid >> 5;
    const int wm   = (wid & 3) * 64;
    const int wn   = (wid >> 2) * 64;
    const int lrow = lane & 15;
    const int off  = (lane >> 4) * 8;

    v8f acc[4][4] = {};

    for (int kk = 0; kk < Kd; kk += 32) {
#pragma unroll
        for (int i = 0; i < 4; i++) {
            int c  = tid + (i << 8);
            int r  = c >> 2;
            int kc = (c & 3) << 3;
            *reinterpret_cast<uint4*>(&As[r][kc]) =
                *reinterpret_cast<const uint4*>(A + (size_t)(m0 + r) * Kd + kk + kc);
        }
#pragma unroll
        for (int i = 0; i < 2; i++) {
            int c  = tid + (i << 8);
            int r  = c >> 2;
            int kc = (c & 3) << 3;
            *reinterpret_cast<uint4*>(&Bs[r][kc]) =
                *reinterpret_cast<const uint4*>(Bt + (size_t)(n0 + r) * Kd + kk + kc);
        }
        if (kk + 32 < Kd)
            __builtin_prefetch(A + (size_t)(m0 + tid) * Kd + kk + 32, 0, 1);
        __syncthreads();

        FragAB a[4], b[4];
#pragma unroll
        for (int i = 0; i < 4; i++) {
            int r = wm + i * 16 + lrow;
            a[i].q[0] = *reinterpret_cast<const uint4*>(&As[r][off]);
            a[i].q[1] = *reinterpret_cast<const uint4*>(&As[r][off + 16]);
        }
#pragma unroll
        for (int j = 0; j < 4; j++) {
            int r = wn + j * 16 + lrow;
            b[j].q[0] = *reinterpret_cast<const uint4*>(&Bs[r][off]);
            b[j].q[1] = *reinterpret_cast<const uint4*>(&Bs[r][off + 16]);
        }
#pragma unroll
        for (int i = 0; i < 4; i++)
#pragma unroll
            for (int j = 0; j < 4; j++)
                acc[i][j] = __builtin_amdgcn_wmma_f32_16x16x32_bf16(
                    false, a[i].v, false, b[j].v, (short)0, acc[i][j], false, false);
        __syncthreads();
    }

#pragma unroll
    for (int i = 0; i < 4; i++) {
#pragma unroll
        for (int j = 0; j < 4; j++) {
#pragma unroll
            for (int v = 0; v < 8; v++) {
                int gm = m0 + wm + i * 16 + v + ((lane >> 4) << 3);
                int gn = n0 + wn + j * 16 + (lane & 15);
                float cv = acc[i][j][v];
                size_t idx = ob + (size_t)gm * N + gn;
                if (out_bf16)
                    ((__bf16*)Out)[idx] = (__bf16)cv;
                else
                    ((float*)Out)[idx] = cv;
            }
        }
    }
}

// ---------------------------------------------------------------------------
// In-place row softmax over 2048 bf16 scores. One block (256 thr) per row;
// each thread keeps its 8 values in registers -> safe in-place rewrite.
// ---------------------------------------------------------------------------
__global__ __launch_bounds__(256) void softmax2048_inplace(__bf16* __restrict__ Sc) {
    const int S = 2048;
    __shared__ float redmax[8];
    __shared__ float redsum[8];
    __bf16* p = Sc + (size_t)blockIdx.x * S;
    const int t = threadIdx.x;

    float vals[8];
    float mx = -3.0e38f;
#pragma unroll
    for (int i = 0; i < 8; i++) {
        vals[i] = (float)p[t + (i << 8)];
        mx = fmaxf(mx, vals[i]);
    }
#pragma unroll
    for (int o = 16; o; o >>= 1) mx = fmaxf(mx, __shfl_xor(mx, o, 32));
    if ((t & 31) == 0) redmax[t >> 5] = mx;
    __syncthreads();
    mx = redmax[0];
#pragma unroll
    for (int i = 1; i < 8; i++) mx = fmaxf(mx, redmax[i]);

    float sm = 0.f;
#pragma unroll
    for (int i = 0; i < 8; i++) {
        vals[i] = __expf(vals[i] - mx);
        sm += vals[i];
    }
#pragma unroll
    for (int o = 16; o; o >>= 1) sm += __shfl_xor(sm, o, 32);
    if ((t & 31) == 0) redsum[t >> 5] = sm;
    __syncthreads();
    sm = 0.f;
#pragma unroll
    for (int i = 0; i < 8; i++) sm += redsum[i];
    const float inv = 1.0f / sm;
#pragma unroll
    for (int i = 0; i < 8; i++) p[t + (i << 8)] = (__bf16)(vals[i] * inv);
}

// ---------------------------------------------------------------------------
extern "C" void kernel_launch(void* const* d_in, const int* in_sizes, int n_in,
                              void* d_out, int out_size, void* d_ws, size_t ws_size,
                              hipStream_t stream) {
    (void)in_sizes; (void)n_in; (void)out_size; (void)ws_size;
    const float* x  = (const float*)d_in[0];
    const float* Wq = (const float*)d_in[1];
    const float* bq = (const float*)d_in[2];
    const float* Wk = (const float*)d_in[3];
    const float* bk = (const float*)d_in[4];
    const float* Wv = (const float*)d_in[5];
    const float* bv = (const float*)d_in[6];
    float* out = (float*)d_out;

    const int B = 4, S = 2048, D = 1024, M = B * S;

    char* ws = (char*)d_ws;
    size_t off = 0;
    auto alloc = [&](size_t bytes) -> void* {
        void* p = ws + off;
        off += (bytes + 255) & ~(size_t)255;
        return p;
    };
    __bf16* Xb  = (__bf16*)alloc((size_t)M * D * 2);  // x in bf16
    __bf16* Wqb = (__bf16*)alloc((size_t)D * D * 2);
    __bf16* Wkb = (__bf16*)alloc((size_t)D * D * 2);
    __bf16* Wvb = (__bf16*)alloc((size_t)D * D * 2);
    __bf16* Qb  = (__bf16*)alloc((size_t)M * D * 2);  // Q pre-scaled by 1/32
    __bf16* Kb  = (__bf16*)alloc((size_t)M * D * 2);
    __bf16* Vt  = (__bf16*)alloc((size_t)M * D * 2);  // [B][D][S]
    __bf16* Sc  = (__bf16*)alloc((size_t)B * S * S * 2);  // scores -> probs in place

    cvt_f32_to_bf16<<<2048, 256, 0, stream>>>(x, Xb, M * D);
    cvt_f32_to_bf16<<<512, 256, 0, stream>>>(Wq, Wqb, D * D);
    cvt_f32_to_bf16<<<512, 256, 0, stream>>>(Wk, Wkb, D * D);
    cvt_f32_to_bf16<<<512, 256, 0, stream>>>(Wv, Wvb, D * D);

    // Q/K/V projections (z = 0/1/2)
    qkv_gemm_wmma<<<dim3(D / 128, M / 256, 3), 256, 0, stream>>>(
        Xb, Wqb, Wkb, Wvb, bq, bk, bv, Qb, Kb, Vt);

    // scores = Q K^T (scale folded into Q), bf16 output
    gemm_bt_wmma<<<dim3(S / 128, S / 256, B), 256, 0, stream>>>(
        Qb, Kb, Sc, S, S, D,
        (long long)S * D, (long long)S * D, (long long)S * S, 1);

    // row softmax in place
    softmax2048_inplace<<<B * S, 256, 0, stream>>>(Sc);

    // out = P V   (B operand is Vt rows = output dim, ld = S)
    gemm_bt_wmma<<<dim3(D / 128, S / 256, B), 256, 0, stream>>>(
        Sc, Vt, out, S, D, S,
        (long long)S * S, (long long)D * S, (long long)S * D, 0);
}
